// Bert_79817672229402
// MI455X (gfx1250) — compile-verified, matrix operand order checked by reading, not run on hardware
//
#include <hip/hip_runtime.h>
#include <math.h>
#include <stdint.h>

// ---------------------------------------------------------------------------
// Problem constants (from reference)
// ---------------------------------------------------------------------------
#define SQ   512          // sequence length S
#define BB   8            // batch B
#define HD   768          // hidden H
#define NHD  12           // heads
#define DH   64           // head dim (q/k)
#define DV   192          // head dim (v)
#define II   2304         // intermediate I
#define LL   4            // layers
#define RR   63           // 2*BK-1 relative buckets
#define MM   (SQ*BB)      // 4096 token rows
#define BH   (BB*NHD)     // 96 (b,h) pairs
#define EPSV 1e-7f
#define SCL  0.072168783649f   // 1/sqrt(3*64)

// LDS staging geometry for the cooperative GEMM
#define KSTEP 32
#define LDSPITCH 40        // 32 + 8 pad elements -> conflict-free fragment reads

typedef __attribute__((ext_vector_type(16))) __bf16 v16bf;
typedef __attribute__((ext_vector_type(8)))  __bf16 v8bf;
typedef __attribute__((ext_vector_type(8)))  float  v8f;

static __device__ __forceinline__ unsigned short f2bf(float f) {
  unsigned int u = __float_as_uint(f);
  u = (u + 0x7FFFu + ((u >> 16) & 1u)) >> 16;   // round-to-nearest-even
  return (unsigned short)u;
}
static __device__ __forceinline__ float geluf(float v) {
  return 0.5f * v * (1.0f + erff(v * 0.70710678118654752f));
}
static __device__ __forceinline__ float block_sum(float v, float* red) {
  int t = threadIdx.x;
  red[t] = v; __syncthreads();
  for (int o = 128; o > 0; o >>= 1) { if (t < o) red[t] += red[t + o]; __syncthreads(); }
  float r = red[0]; __syncthreads();
  return r;
}
static __device__ __forceinline__ unsigned lds_off(const void* p) {
  return (unsigned)(unsigned long long)(uintptr_t)p;   // low 32 bits = LDS offset
}
// one 16-byte async global->LDS copy per lane
static __device__ __forceinline__ void async_copy16(unsigned ldsDst, const void* gSrc) {
  asm volatile("global_load_async_to_lds_b128 %0, %1, off"
               :: "v"(ldsDst), "v"((unsigned long long)(uintptr_t)gSrc)
               : "memory");
}
static __device__ __forceinline__ void wait_async0() {
  asm volatile("s_wait_asynccnt 0x0" ::: "memory");
}

// ---------------------------------------------------------------------------
// f32 -> bf16 cast
// ---------------------------------------------------------------------------
__global__ void cast_bf16_kernel(const float* __restrict__ src,
                                 unsigned short* __restrict__ dst, long long n) {
  long long t = (long long)blockIdx.x * blockDim.x + threadIdx.x;
  if (t < n) dst[t] = f2bf(src[t]);
}

// ---------------------------------------------------------------------------
// embedding gather + LayerNorm  (one block per token row, H=768)
// ---------------------------------------------------------------------------
__global__ __launch_bounds__(256) void embed_ln_kernel(
    const int* __restrict__ ids, const float* __restrict__ wemb,
    float* __restrict__ x) {
  __shared__ float buf[HD];
  __shared__ float red[256];
  int m = blockIdx.x;
  const float* src = wemb + (long long)ids[m] * HD;
  float s = 0.f;
  for (int i = threadIdx.x; i < HD; i += 256) { float v = src[i]; buf[i] = v; s += v; }
  s = block_sum(s, red);
  float mean = s / (float)HD;
  float vs = 0.f;
  for (int i = threadIdx.x; i < HD; i += 256) { float d = buf[i] - mean; vs += d * d; }
  vs = block_sum(vs, red);
  float inv = rsqrtf(vs / (float)HD + EPSV);
  float* dst = x + (long long)m * HD;
  for (int i = threadIdx.x; i < HD; i += 256) dst[i] = (buf[i] - mean) * inv;
}

// rel = LN(rel_emb) * w + b   (63 rows)
__global__ __launch_bounds__(256) void rel_ln_kernel(
    const float* __restrict__ remb, const float* __restrict__ w,
    const float* __restrict__ b, float* __restrict__ rel) {
  __shared__ float buf[HD];
  __shared__ float red[256];
  int r = blockIdx.x;
  const float* src = remb + (long long)r * HD;
  float s = 0.f;
  for (int i = threadIdx.x; i < HD; i += 256) { float v = src[i]; buf[i] = v; s += v; }
  s = block_sum(s, red);
  float mean = s / (float)HD;
  float vs = 0.f;
  for (int i = threadIdx.x; i < HD; i += 256) { float d = buf[i] - mean; vs += d * d; }
  vs = block_sum(vs, red);
  float inv = rsqrtf(vs / (float)HD + EPSV);
  float* dst = rel + (long long)r * HD;
  for (int i = threadIdx.x; i < HD; i += 256)
    dst[i] = (buf[i] - mean) * inv * w[i] + b[i];
}

// h = LN(x) cast to bf16
__global__ __launch_bounds__(256) void ln_cast_kernel(
    const float* __restrict__ x, unsigned short* __restrict__ hb) {
  __shared__ float buf[HD];
  __shared__ float red[256];
  int m = blockIdx.x;
  const float* src = x + (long long)m * HD;
  float s = 0.f;
  for (int i = threadIdx.x; i < HD; i += 256) { float v = src[i]; buf[i] = v; s += v; }
  s = block_sum(s, red);
  float mean = s / (float)HD;
  float vs = 0.f;
  for (int i = threadIdx.x; i < HD; i += 256) { float d = buf[i] - mean; vs += d * d; }
  vs = block_sum(vs, red);
  float inv = rsqrtf(vs / (float)HD + EPSV);
  unsigned short* dst = hb + (long long)m * HD;
  for (int i = threadIdx.x; i < HD; i += 256) dst[i] = f2bf((buf[i] - mean) * inv);
}

// ---------------------------------------------------------------------------
// Cooperative WMMA GEMM with async global->LDS staging (CDNA5 path):
//   C[M,N](f32) = A[M,K](bf16) * B[N,K](bf16)^T  (+bias[n]) (+Cadd[m,n])
// Block = 256 threads (8 waves, 2 M-waves x 4 N-waves) -> 128x128 tile.
// K staged in 32-steps through double-buffered LDS filled with
// global_load_async_to_lds_b128 (ASYNCcnt), fragments read from LDS.
// Requires M%128==0, N%128==0, K%32==0.
// ---------------------------------------------------------------------------
__global__ __launch_bounds__(256) void wmma_gemm_staged_kernel(
    const unsigned short* __restrict__ A, const unsigned short* __restrict__ Bm,
    float* __restrict__ C, const float* __restrict__ bias,
    const float* __restrict__ Cadd, int M, int N, int K) {
  __shared__ __align__(16) unsigned short Asm[2][128 * LDSPITCH];
  __shared__ __align__(16) unsigned short Bsm[2][128 * LDSPITCH];

  const int n0 = blockIdx.x * 128;
  const int m0 = blockIdx.y * 128;
  const int t = threadIdx.x;
  const int wave = t >> 5, lane = t & 31;
  const int lm = lane & 15, hi = lane >> 4;
  const int wm = wave & 1;        // M half (64 rows)
  const int wn = wave >> 1;       // N quarter (32 cols)

  // per-thread staging chunks: 2 x 16B for A, 2 x 16B for B per stage
  const int c0 = t * 2, c1 = t * 2 + 1;
  const int ar0 = c0 >> 2, ac0 = (c0 & 3) * 8;
  const int ar1 = c1 >> 2, ac1 = (c1 & 3) * 8;

  v8f acc[4][2] = {};
  const int nstages = K / KSTEP;
  int cur = 0;

  // prologue: stage 0
  {
    const int ks = 0;
    async_copy16(lds_off(&Asm[0][ar0 * LDSPITCH + ac0]),
                 A + (long long)(m0 + ar0) * K + ks + ac0);
    async_copy16(lds_off(&Asm[0][ar1 * LDSPITCH + ac1]),
                 A + (long long)(m0 + ar1) * K + ks + ac1);
    async_copy16(lds_off(&Bsm[0][ar0 * LDSPITCH + ac0]),
                 Bm + (long long)(n0 + ar0) * K + ks + ac0);
    async_copy16(lds_off(&Bsm[0][ar1 * LDSPITCH + ac1]),
                 Bm + (long long)(n0 + ar1) * K + ks + ac1);
    wait_async0();
    __syncthreads();
  }

  for (int s = 0; s < nstages; s++) {
    // prefetch next stage into the alternate buffer
    if (s + 1 < nstages) {
      const int ks = (s + 1) * KSTEP;
      const int nxt = cur ^ 1;
      async_copy16(lds_off(&Asm[nxt][ar0 * LDSPITCH + ac0]),
                   A + (long long)(m0 + ar0) * K + ks + ac0);
      async_copy16(lds_off(&Asm[nxt][ar1 * LDSPITCH + ac1]),
                   A + (long long)(m0 + ar1) * K + ks + ac1);
      async_copy16(lds_off(&Bsm[nxt][ar0 * LDSPITCH + ac0]),
                   Bm + (long long)(n0 + ar0) * K + ks + ac0);
      async_copy16(lds_off(&Bsm[nxt][ar1 * LDSPITCH + ac1]),
                   Bm + (long long)(n0 + ar1) * K + ks + ac1);
    }

    // consume current stage from LDS
    v16bf a[4], b[2];
#pragma unroll
    for (int i = 0; i < 4; i++) {
      const __bf16* p =
          (const __bf16*)&Asm[cur][(wm * 64 + 16 * i + lm) * LDSPITCH + hi * 8];
      v8bf lo = *(const v8bf*)p;
      v8bf hh = *(const v8bf*)(p + 16);
#pragma unroll
      for (int e = 0; e < 8; e++) { a[i][e] = lo[e]; a[i][8 + e] = hh[e]; }
    }
#pragma unroll
    for (int j = 0; j < 2; j++) {
      const __bf16* p =
          (const __bf16*)&Bsm[cur][(wn * 32 + 16 * j + lm) * LDSPITCH + hi * 8];
      v8bf lo = *(const v8bf*)p;
      v8bf hh = *(const v8bf*)(p + 16);
#pragma unroll
      for (int e = 0; e < 8; e++) { b[j][e] = lo[e]; b[j][8 + e] = hh[e]; }
    }
#pragma unroll
    for (int i = 0; i < 4; i++)
#pragma unroll
      for (int j = 0; j < 2; j++)
        acc[i][j] = __builtin_amdgcn_wmma_f32_16x16x32_bf16(
            false, a[i], false, b[j], (short)0, acc[i][j], false, false);

    wait_async0();      // this wave's prefetch landed
    __syncthreads();    // everyone's prefetch landed / tile consumed
    cur ^= 1;
  }

#pragma unroll
  for (int i = 0; i < 4; i++)
#pragma unroll
    for (int j = 0; j < 2; j++) {
      const int n = n0 + wn * 32 + 16 * j + lm;
      const float bv = bias ? bias[n] : 0.f;
#pragma unroll
      for (int r = 0; r < 8; r++) {
        const int m = m0 + wm * 64 + 16 * i + r + hi * 8;
        const long long idx = (long long)m * N + n;
        float v = acc[i][j][r] + bv;
        if (Cadd) v += Cadd[idx];
        C[idx] = v;
      }
    }
}

// ---------------------------------------------------------------------------
// Direct (non-staged) WMMA GEMM, 1 wave -> 64x64 tile; used for batched P.V
// (N=192 is not a multiple of 128). Batched via gridDim.z + element strides.
// ---------------------------------------------------------------------------
__global__ __launch_bounds__(32) void wmma_gemm_kernel(
    const unsigned short* __restrict__ A, const unsigned short* __restrict__ Bm,
    float* __restrict__ C, int M, int N, int K,
    long long sA, long long sB, long long sC) {
  long long batch = blockIdx.z;
  A  += batch * sA;
  Bm += batch * sB;
  C  += batch * sC;
  const int n0 = blockIdx.x * 64;
  const int m0 = blockIdx.y * 64;
  const int lane = threadIdx.x & 31;
  const int lm = lane & 15;
  const int hi = lane >> 4;

  v8f acc[4][4] = {};
  for (int kk = 0; kk < K; kk += 32) {
    const int kb = kk + hi * 8;
    v16bf a[4], b[4];
#pragma unroll
    for (int i = 0; i < 4; i++) {
      const __bf16* p = (const __bf16*)(A + (long long)(m0 + 16 * i + lm) * K + kb);
      v8bf lo = *(const v8bf*)p;
      v8bf hh = *(const v8bf*)(p + 16);
#pragma unroll
      for (int e = 0; e < 8; e++) { a[i][e] = lo[e]; a[i][8 + e] = hh[e]; }
    }
#pragma unroll
    for (int j = 0; j < 4; j++) {
      const __bf16* p = (const __bf16*)(Bm + (long long)(n0 + 16 * j + lm) * K + kb);
      v8bf lo = *(const v8bf*)p;
      v8bf hh = *(const v8bf*)(p + 16);
#pragma unroll
      for (int e = 0; e < 8; e++) { b[j][e] = lo[e]; b[j][8 + e] = hh[e]; }
    }
#pragma unroll
    for (int i = 0; i < 4; i++)
#pragma unroll
      for (int j = 0; j < 4; j++)
        acc[i][j] = __builtin_amdgcn_wmma_f32_16x16x32_bf16(
            false, a[i], false, b[j], (short)0, acc[i][j], false, false);
  }
#pragma unroll
  for (int i = 0; i < 4; i++)
#pragma unroll
    for (int j = 0; j < 4; j++) {
      const int n = n0 + 16 * j + lm;
#pragma unroll
      for (int r = 0; r < 8; r++) {
        const int m = m0 + 16 * i + r + hi * 8;
        C[(long long)m * N + n] = acc[i][j][r];
      }
    }
}

// ---------------------------------------------------------------------------
// pos[r,o] = rel[r,:] . Wqk[o,:] + bqk[o]    (63 x 1536, K=768 -- tiny)
// ---------------------------------------------------------------------------
__global__ void pos_proj_kernel(const float* __restrict__ rel,
                                const float* __restrict__ Wqk,
                                const float* __restrict__ bqk,
                                float* __restrict__ pos) {
  int t = blockIdx.x * 256 + threadIdx.x;
  if (t >= RR * 2 * HD) return;
  int r = t / (2 * HD), o = t % (2 * HD);
  const float* a = rel + (long long)r * HD;
  const float* w = Wqk + (long long)o * HD;
  float s = bqk[o];
  for (int k = 0; k < HD; k++) s += a[k] * w[k];
  pos[t] = s;
}

// out[(bh*S + s)*63 + r] = sum_d qk[(s*B+b)*1536 + xoff + h*64 + d] *
//                                pos[r*1536 + poff + h*64 + d]
__global__ void posdot_kernel(const float* __restrict__ qkbuf,
                              const float* __restrict__ pos,
                              float* __restrict__ out, int xoff, int poff) {
  long long t = (long long)blockIdx.x * 256 + threadIdx.x;
  if (t >= (long long)BH * SQ * RR) return;
  int r = (int)(t % RR);
  int s = (int)((t / RR) % SQ);
  int bh = (int)(t / ((long long)RR * SQ));
  int b = bh / NHD, h = bh % NHD;
  const float* xp = qkbuf + ((long long)(s * BB + b)) * (2 * HD) + xoff + h * DH;
  const float* pp = pos + (long long)r * (2 * HD) + poff + h * DH;
  float acc = 0.f;
  for (int d = 0; d < DH; d++) acc += xp[d] * pp[d];
  out[t] = acc;
}

// ---------------------------------------------------------------------------
// Repack q,k,v into per-(b,h) bf16 tiles:
//   qb/kb : [bh][s][64],  vT : [bh][dv][s]  (transposed for P.V GEMM)
// ---------------------------------------------------------------------------
__global__ void qkv_cast_kernel(const float* __restrict__ qk,
                                const float* __restrict__ vg,
                                unsigned short* __restrict__ qb,
                                unsigned short* __restrict__ kb,
                                unsigned short* __restrict__ vT) {
  long long t = (long long)blockIdx.x * 256 + threadIdx.x;
  const long long NQ = (long long)BH * SQ * DH;
  if (t < NQ) {
    int d = (int)(t & 63);
    int s = (int)((t >> 6) & (SQ - 1));
    int bh = (int)(t >> 15);
    int b = bh / NHD, h = bh % NHD;
    long long row = (long long)(s * BB + b) * (2 * HD);
    qb[t] = f2bf(qk[row + h * DH + d]);
    kb[t] = f2bf(qk[row + HD + h * DH + d]);
  } else {
    long long u = t - NQ;
    if (u >= (long long)BH * DV * SQ) return;
    int s = (int)(u % SQ);
    long long dvbh = u / SQ;
    int dv = (int)(dvbh % DV);
    int bh = (int)(dvbh / DV);
    int b = bh / NHD, h = bh % NHD;
    vT[u] = f2bf(vg[(long long)(s * BB + b) * (2 * II) + h * DV + dv]);
  }
}

// ---------------------------------------------------------------------------
// Attention scores: WMMA Q.K^T for a 16-row q-tile across all 512 keys,
// + gathered positional dot terms, softmax, write P (bf16).
// Block = 256 threads (8 waves); wave w owns key columns [w*64, w*64+64).
// ---------------------------------------------------------------------------
__global__ __launch_bounds__(256) void attn_scores_kernel(
    const unsigned short* __restrict__ qb, const unsigned short* __restrict__ kb,
    const float* __restrict__ qdot, const float* __restrict__ kdot,
    const int* __restrict__ pidx, unsigned short* __restrict__ P) {
  __shared__ float sc[16][SQ];
  const int bh = blockIdx.y;
  const int q0 = blockIdx.x * 16;
  const int wave = threadIdx.x >> 5, lane = threadIdx.x & 31;
  const int lm = lane & 15, hi = lane >> 4;
  const int n0 = wave * 64;
  const unsigned short* Q  = qb + (long long)bh * SQ * DH;
  const unsigned short* Kk = kb + (long long)bh * SQ * DH;

  v8f acc[4] = {};
  for (int kk = 0; kk < DH; kk += 32) {
    const int kbase = kk + hi * 8;
    v16bf a;
    {
      const __bf16* p = (const __bf16*)(Q + (long long)(q0 + lm) * DH + kbase);
      v8bf lo = *(const v8bf*)p;
      v8bf hh = *(const v8bf*)(p + 16);
#pragma unroll
      for (int e = 0; e < 8; e++) { a[e] = lo[e]; a[8 + e] = hh[e]; }
    }
#pragma unroll
    for (int j = 0; j < 4; j++) {
      const __bf16* p = (const __bf16*)(Kk + (long long)(n0 + 16 * j + lm) * DH + kbase);
      v8bf lo = *(const v8bf*)p;
      v8bf hh = *(const v8bf*)(p + 16);
      v16bf b;
#pragma unroll
      for (int e = 0; e < 8; e++) { b[e] = lo[e]; b[8 + e] = hh[e]; }
      acc[j] = __builtin_amdgcn_wmma_f32_16x16x32_bf16(
          false, a, false, b, (short)0, acc[j], false, false);
    }
  }
#pragma unroll
  for (int j = 0; j < 4; j++)
#pragma unroll
    for (int r = 0; r < 8; r++)
      sc[r + hi * 8][n0 + 16 * j + lm] = acc[j][r];
  __syncthreads();

  // softmax: 8 waves x 2 rows = 16 rows; each lane covers 16 columns
  for (int rr = 0; rr < 2; rr++) {
    const int lr = wave * 2 + rr;
    const int q = q0 + lr;
    const float* qd = qdot + ((long long)bh * SQ + q) * RR;
    float vals[16];
    float mx = -3.0e38f;
#pragma unroll
    for (int it = 0; it < 16; it++) {
      const int c = lane + it * 32;
      const int r = pidx[q * SQ + c];
      float s = (sc[lr][c] + qd[r] + kdot[((long long)bh * SQ + c) * RR + r]) * SCL;
      vals[it] = s;
      mx = fmaxf(mx, s);
    }
    for (int o = 16; o > 0; o >>= 1) mx = fmaxf(mx, __shfl_xor(mx, o, 32));
    float sum = 0.f;
#pragma unroll
    for (int it = 0; it < 16; it++) { vals[it] = __expf(vals[it] - mx); sum += vals[it]; }
    for (int o = 16; o > 0; o >>= 1) sum += __shfl_xor(sum, o, 32);
    const float inv = 1.0f / sum;
    unsigned short* Pp = P + ((long long)bh * SQ + q) * SQ;
#pragma unroll
    for (int it = 0; it < 16; it++) Pp[lane + it * 32] = f2bf(vals[it] * inv);
  }
}

// ---------------------------------------------------------------------------
// GLU epilogue + LayerNorm over I=2304:
//   c = ctx + sigmoid(l_skip)*gelu(value);  c *= gelu(gate);  LN -> bf16
// ---------------------------------------------------------------------------
__global__ __launch_bounds__(256) void glu_ln_kernel(
    const float* __restrict__ ctxbh, const float* __restrict__ vg,
    const float* __restrict__ lskip, unsigned short* __restrict__ ctxn) {
  __shared__ float buf[II];
  __shared__ float red[256];
  const int m = blockIdx.x;
  const int b = m % BB, s = m / BB;
  float sum = 0.f;
  for (int n = threadIdx.x; n < II; n += 256) {
    const int hh = n / DV, dv = n % DV;
    float c = ctxbh[((long long)(b * NHD + hh) * SQ + s) * DV + dv];
    const float value = vg[(long long)m * (2 * II) + n];
    const float gate  = vg[(long long)m * (2 * II) + II + n];
    c += (1.f / (1.f + __expf(-lskip[n]))) * geluf(value);
    c *= geluf(gate);
    buf[n] = c;
    sum += c;
  }
  sum = block_sum(sum, red);
  const float mean = sum / (float)II;
  float vs = 0.f;
  for (int n = threadIdx.x; n < II; n += 256) { float d = buf[n] - mean; vs += d * d; }
  vs = block_sum(vs, red);
  const float inv = rsqrtf(vs / (float)II + EPSV);
  for (int n = threadIdx.x; n < II; n += 256)
    ctxn[(long long)m * II + n] = f2bf((buf[n] - mean) * inv);
}

__global__ void copy_kernel(const float* __restrict__ src, float* __restrict__ dst,
                            long long n) {
  long long t = (long long)blockIdx.x * 256 + threadIdx.x;
  if (t < n) dst[t] = src[t];
}

// ---------------------------------------------------------------------------
// Host-side orchestration
// ---------------------------------------------------------------------------
extern "C" void kernel_launch(void* const* d_in, const int* in_sizes, int n_in,
                              void* d_out, int out_size, void* d_ws, size_t ws_size,
                              hipStream_t stream) {
  const int*   ids   = (const int*)d_in[0];
  const int*   pidx  = (const int*)d_in[2];
  const float* wemb  = (const float*)d_in[3];
  const float* remb  = (const float*)d_in[4];
  const float* rlw   = (const float*)d_in[5];
  const float* rlb   = (const float*)d_in[6];
  const float* Wv    = (const float*)d_in[7];   // [L, 2I, H]
  const float* Wqk   = (const float*)d_in[8];   // [L, 2H, H]
  const float* bqk   = (const float*)d_in[9];   // [L, 2H]
  const float* Wo    = (const float*)d_in[10];  // [L, H, I]
  const float* lskip = (const float*)d_in[11];  // [L, I]

  size_t off = 0;
  auto alloc = [&](size_t bytes) -> void* {
    off = (off + 255) & ~(size_t)255;
    void* p = (char*)d_ws + off;
    off += bytes;
    return p;
  };
  float*          xf32   = (float*)alloc((size_t)MM * HD * 4);
  unsigned short* hb     = (unsigned short*)alloc((size_t)MM * HD * 2);
  float*          relf   = (float*)alloc((size_t)RR * HD * 4);
  unsigned short* Wvb    = (unsigned short*)alloc((size_t)2 * II * HD * 2);
  unsigned short* Wqkb   = (unsigned short*)alloc((size_t)2 * HD * HD * 2);
  unsigned short* Wob    = (unsigned short*)alloc((size_t)HD * II * 2);
  float*          vg     = (float*)alloc((size_t)MM * 2 * II * 4);
  float*          qk     = (float*)alloc((size_t)MM * 2 * HD * 4);
  float*          posb   = (float*)alloc((size_t)RR * 2 * HD * 4);
  float*          qdot   = (float*)alloc((size_t)BH * SQ * RR * 4);
  float*          kdot   = (float*)alloc((size_t)BH * SQ * RR * 4);
  unsigned short* qb     = (unsigned short*)alloc((size_t)BH * SQ * DH * 2);
  unsigned short* kb     = (unsigned short*)alloc((size_t)BH * SQ * DH * 2);
  unsigned short* vT     = (unsigned short*)alloc((size_t)BH * DV * SQ * 2);
  unsigned short* P      = (unsigned short*)alloc((size_t)BH * SQ * SQ * 2);
  float*          ctxbh  = (float*)alloc((size_t)BH * SQ * DV * 4);
  unsigned short* ctxn   = (unsigned short*)alloc((size_t)MM * II * 2);
  (void)ws_size; (void)n_in; (void)in_sizes; (void)out_size;

  // initial embedding LN and relative-embedding LN
  embed_ln_kernel<<<MM, 256, 0, stream>>>(ids, wemb, xf32);
  rel_ln_kernel<<<RR, 256, 0, stream>>>(remb, rlw, rlb, relf);

  for (int l = 0; l < LL; l++) {
    const float* Wv_l   = Wv  + (long long)l * 2 * II * HD;
    const float* Wqk_l  = Wqk + (long long)l * 2 * HD * HD;
    const float* bqk_l  = bqk + (long long)l * 2 * HD;
    const float* Wo_l   = Wo  + (long long)l * HD * II;
    const float* lsk_l  = lskip + (long long)l * II;

    // cast weights to bf16
    {
      long long n1 = (long long)2 * II * HD;
      cast_bf16_kernel<<<(n1 + 255) / 256, 256, 0, stream>>>(Wv_l, Wvb, n1);
      long long n2 = (long long)2 * HD * HD;
      cast_bf16_kernel<<<(n2 + 255) / 256, 256, 0, stream>>>(Wqk_l, Wqkb, n2);
      long long n3 = (long long)HD * II;
      cast_bf16_kernel<<<(n3 + 255) / 256, 256, 0, stream>>>(Wo_l, Wob, n3);
    }

    // h = LN(x) -> bf16
    ln_cast_kernel<<<MM, 256, 0, stream>>>(xf32, hb);

    // vg = h . Wv^T   [4096 x 4608], K=768  (async-staged WMMA)
    {
      dim3 grid(2 * II / 128, MM / 128, 1);
      wmma_gemm_staged_kernel<<<grid, 256, 0, stream>>>(hb, Wvb, vg, nullptr,
                                                        nullptr, MM, 2 * II, HD);
    }
    // qk = h . Wqk^T + bqk   [4096 x 1536], K=768
    {
      dim3 grid(2 * HD / 128, MM / 128, 1);
      wmma_gemm_staged_kernel<<<grid, 256, 0, stream>>>(hb, Wqkb, qk, bqk_l,
                                                        nullptr, MM, 2 * HD, HD);
    }
    // pos = rel . Wqk^T + bqk   [63 x 1536]
    {
      int n = RR * 2 * HD;
      pos_proj_kernel<<<(n + 255) / 256, 256, 0, stream>>>(relf, Wqk_l, bqk_l, posb);
    }
    // positional dot tables
    {
      long long n = (long long)BH * SQ * RR;
      int g = (int)((n + 255) / 256);
      posdot_kernel<<<g, 256, 0, stream>>>(qk, posb, qdot, 0, HD);   // q . k_pos
      posdot_kernel<<<g, 256, 0, stream>>>(qk, posb, kdot, HD, 0);   // k . q_pos
    }
    // repack q/k/v
    {
      long long n = (long long)BH * SQ * DH + (long long)BH * DV * SQ;
      qkv_cast_kernel<<<(int)((n + 255) / 256), 256, 0, stream>>>(qk, vg, qb, kb, vT);
    }
    // scores + softmax -> P
    {
      dim3 grid(SQ / 16, BH, 1);
      attn_scores_kernel<<<grid, 256, 0, stream>>>(qb, kb, qdot, kdot, pidx, P);
    }
    // ctx = P . V   batched over 96 (b,h):  [512 x 192], K=512
    {
      dim3 grid(DV / 64, SQ / 64, BH);
      wmma_gemm_kernel<<<grid, 32, 0, stream>>>(
          P, vT, ctxbh, SQ, DV, SQ,
          (long long)SQ * SQ, (long long)DV * SQ, (long long)SQ * DV);
    }
    // GLU epilogue + LN -> ctxn (bf16)
    glu_ln_kernel<<<MM, 256, 0, stream>>>(ctxbh, vg, lsk_l, ctxn);

    // x = x + ctxn . Wo^T   [4096 x 768], K=2304  (async-staged WMMA)
    {
      dim3 grid(HD / 128, MM / 128, 1);
      wmma_gemm_staged_kernel<<<grid, 256, 0, stream>>>(ctxn, Wob, xf32, nullptr,
                                                        xf32, MM, HD, II);
    }
  }

  {
    long long n = (long long)MM * HD;
    copy_kernel<<<(int)((n + 255) / 256), 256, 0, stream>>>(xf32, (float*)d_out, n);
  }
}
